// Model_17944373363210
// MI455X (gfx1250) — matrix operand order repaired and checked by reading, not compiled, not run on hardware
//
#include <hip/hip_runtime.h>
#include <hip/hip_bf16.h>
#include <math.h>

typedef _Float16 f16;
typedef __attribute__((ext_vector_type(8)))  _Float16 v8h;
typedef __attribute__((ext_vector_type(16))) _Float16 v16h;
typedef __attribute__((ext_vector_type(8)))  float    v8f;
typedef __attribute__((ext_vector_type(4)))  int      v4i;

#define BATCH 8
#define SEQL  2048
#define CIN   21
#define DMODEL 512
#define DINNER 1024
#define DSTATE 16
#define DTRANK 32
#define MARKD  4
#define PREDL  96
#define MROWS (BATCH * SEQL)   // 16384

// ---------------------------------------------------------------------------
// CDNA5 async global->LDS path (guarded; falls back to sync staging)
// ---------------------------------------------------------------------------
#if __has_builtin(__builtin_amdgcn_global_load_async_to_lds_b128)
#define USE_ASYNC_LDS 1
#else
#define USE_ASYNC_LDS 0
#endif

template <int N>
__device__ __forceinline__ void async_wait() {
#if __has_builtin(__builtin_amdgcn_s_wait_asynccnt)
    __builtin_amdgcn_s_wait_asynccnt(N);
#else
    asm volatile("s_wait_asynccnt %0" ::"i"(N) : "memory");
#endif
}

#if USE_ASYNC_LDS
typedef __attribute__((address_space(1))) v4i gv4i;
typedef __attribute__((address_space(3))) v4i lv4i;
__device__ __forceinline__ void async_b128(const void* g, void* l) {
    __builtin_amdgcn_global_load_async_to_lds_b128((gv4i*)g, (lv4i*)l, 0, 0);
}
#endif

// ---------------------------------------------------------------------------
// generic f32 -> f16 convert (weights)
// ---------------------------------------------------------------------------
__global__ void cvt_f32_f16_kernel(const float* __restrict__ in,
                                   f16* __restrict__ out, size_t n) {
    for (size_t i = (size_t)blockIdx.x * blockDim.x + threadIdx.x; i < n;
         i += (size_t)gridDim.x * blockDim.x)
        out[i] = (f16)in[i];
}

// ---------------------------------------------------------------------------
// RevIN stats per (b, c): mean, std, 1/std
// ---------------------------------------------------------------------------
__global__ __launch_bounds__(256) void stats_kernel(
    const float* __restrict__ x, float* __restrict__ mean,
    float* __restrict__ istd, float* __restrict__ stdv) {
    int bc = blockIdx.x;                // 0..167
    int b = bc / CIN, c = bc % CIN;
    float s = 0.f, sq = 0.f;
    for (int l = threadIdx.x; l < SEQL; l += 256) {
        float v = x[((size_t)b * SEQL + l) * CIN + c];
        s += v; sq += v * v;
    }
    __shared__ float ss[256], sqq[256];
    ss[threadIdx.x] = s; sqq[threadIdx.x] = sq;
    __syncthreads();
    for (int st = 128; st > 0; st >>= 1) {
        if (threadIdx.x < st) {
            ss[threadIdx.x]  += ss[threadIdx.x + st];
            sqq[threadIdx.x] += sqq[threadIdx.x + st];
        }
        __syncthreads();
    }
    if (threadIdx.x == 0) {
        float m   = ss[0] / (float)SEQL;
        float var = sqq[0] / (float)SEQL - m * m;
        float sd  = sqrtf(var + 1e-5f);
        mean[bc] = m; stdv[bc] = sd; istd[bc] = 1.f / sd;
    }
}

// ---------------------------------------------------------------------------
// Embedding: circular conv1d(k=3) token embed + time-feature embed + pos embed
// one block per (b,l), 128 threads over d
// ---------------------------------------------------------------------------
__global__ __launch_bounds__(128) void embed_kernel(
    const float* __restrict__ x, const float* __restrict__ mark,
    const float* __restrict__ conv_w, const float* __restrict__ temp_w,
    const float* __restrict__ mean, const float* __restrict__ istd,
    f16* __restrict__ emb) {
    int bl = blockIdx.x;
    int b = bl >> 11, l = bl & (SEQL - 1);
    __shared__ float sx[3][CIN];
    __shared__ float sm[MARKD];
    int tid = threadIdx.x;
    if (tid < CIN) {
        int lm = (l + SEQL - 1) & (SEQL - 1);
        int lp = (l + 1) & (SEQL - 1);
        float mu = mean[b * CIN + tid], is = istd[b * CIN + tid];
        sx[0][tid] = (x[((size_t)b * SEQL + lm) * CIN + tid] - mu) * is;
        sx[1][tid] = (x[((size_t)b * SEQL + l ) * CIN + tid] - mu) * is;
        sx[2][tid] = (x[((size_t)b * SEQL + lp) * CIN + tid] - mu) * is;
    }
    if (tid >= 32 && tid < 32 + MARKD)
        sm[tid - 32] = mark[((size_t)b * SEQL + l) * MARKD + (tid - 32)];
    __syncthreads();
    for (int d = tid; d < DMODEL; d += 128) {
        float acc = 0.f;
        for (int c = 0; c < CIN; c++) {
            const float* w = &conv_w[((size_t)d * CIN + c) * 3];
            acc += w[0] * sx[0][c] + w[1] * sx[1][c] + w[2] * sx[2][c];
        }
        #pragma unroll
        for (int j = 0; j < MARKD; j++) acc += temp_w[d * MARKD + j] * sm[j];
        // positional embedding
        int  i2  = d & ~1;
        float dv = __expf(-(float)i2 * (9.210340371976184f / (float)DMODEL));
        acc += (d & 1) ? __cosf((float)l * dv) : __sinf((float)l * dv);
        emb[(size_t)bl * DMODEL + d] = (f16)acc;
    }
}

// ---------------------------------------------------------------------------
// WMMA GEMM: C[M,N] = A[M,K] (f16, row-major) x W[N,K]^T (f16, row-major)
// block tile 64x128, 8 waves (4 in M, 2 in N), wave tile 16x64, K-step 32.
// Double-buffered LDS; tiles staged with GLOBAL_LOAD_ASYNC_TO_LDS_B128
// (ASYNCcnt pipelined: next tile's 3 loads/wave in flight behind the WMMAs).
// MODE 0: store f16   MODE 1: store f32   MODE 2: softplus(acc+bias) -> f32
// ---------------------------------------------------------------------------
#define GBM 64
#define GBN 128
#define GBK 32

template <int MODE>
__global__ __launch_bounds__(256) void gemm_wmma_kernel(
    const f16* __restrict__ A, const f16* __restrict__ W,
    const float* __restrict__ bias, void* __restrict__ Cout,
    int M, int N, int K) {
    __shared__ __align__(32) f16 As2[2][GBM][GBK];   // 2 x 4 KB
    __shared__ __align__(32) f16 Bs2[2][GBN][GBK];   // 2 x 8 KB
    int tid  = threadIdx.x;
    int bm   = blockIdx.y * GBM;
    int bn   = blockIdx.x * GBN;
    int wave = tid >> 5, lane = tid & 31;
    int wm   = (wave & 3) * 16;   // wave M offset in block
    int wn   = (wave >> 2) * 64;  // wave N offset in block
    v8f acc[4] = {};

    int a_row = tid >> 2, a_col = (tid & 3) * 8;
    // B chunks: 128 rows x 4 x 8-f16 chunks = 512 chunks; 2 per thread.
    int b_row0 = tid >> 2,          b_col0 = (tid & 3) * 8;
    int b_row1 = (tid + 256) >> 2,  b_col1 = b_col0;
    // clamp OOB weight rows instead of predicating: async instruction count
    // stays exactly 3/wave/tile, garbage lands only in cols >= N (never stored)
    int gnr0 = bn + b_row0; if (gnr0 > N - 1) gnr0 = N - 1;
    int gnr1 = bn + b_row1; if (gnr1 > N - 1) gnr1 = N - 1;

    int g    = lane >> 4;             // lane K-group
    int mrow = wm + (lane & 15);
    int lnn  = lane & 15;

    const int ktiles = K / GBK;

#if USE_ASYNC_LDS
    // prologue: stage tile 0 into buffer 0
    {
        const int k0 = 0;
        async_b128(&A[(size_t)(bm + a_row) * K + k0 + a_col], &As2[0][a_row][a_col]);
        async_b128(&W[(size_t)gnr0 * K + k0 + b_col0],        &Bs2[0][b_row0][b_col0]);
        async_b128(&W[(size_t)gnr1 * K + k0 + b_col1],        &Bs2[0][b_row1][b_col1]);
    }
#endif

    for (int kt = 0; kt < ktiles; kt++) {
        int buf = kt & 1;
#if USE_ASYNC_LDS
        if (kt + 1 < ktiles) {
            const int k1 = (kt + 1) * GBK;
            async_b128(&A[(size_t)(bm + a_row) * K + k1 + a_col], &As2[buf ^ 1][a_row][a_col]);
            async_b128(&W[(size_t)gnr0 * K + k1 + b_col0],        &Bs2[buf ^ 1][b_row0][b_col0]);
            async_b128(&W[(size_t)gnr1 * K + k1 + b_col1],        &Bs2[buf ^ 1][b_row1][b_col1]);
            async_wait<3>();   // tile kt landed; kt+1's 3 loads stay in flight
        } else {
            async_wait<0>();
        }
#else
        const int k0 = kt * GBK;
        *(v8h*)&As2[buf][a_row][a_col] =
            *(const v8h*)&A[(size_t)(bm + a_row) * K + k0 + a_col];
        *(v8h*)&Bs2[buf][b_row0][b_col0] =
            *(const v8h*)&W[(size_t)gnr0 * K + k0 + b_col0];
        *(v8h*)&Bs2[buf][b_row1][b_col1] =
            *(const v8h*)&W[(size_t)gnr1 * K + k0 + b_col1];
        if (kt + 1 < ktiles)
            __builtin_prefetch(&A[(size_t)(bm + a_row) * K + k0 + GBK + a_col], 0, 1);
#endif
        __syncthreads();

        // A fragment: lane %16 = M row; 16 contiguous K at 16*g
        v16h afrag = *(const v16h*)&As2[buf][mrow][16 * g];
        // B fragments: lane %16 = N col; 16 contiguous K at 16*g
        #pragma unroll
        for (int t = 0; t < 4; t++) {
            int ncol = wn + t * 16 + lnn;
            v16h bfrag = *(const v16h*)&Bs2[buf][ncol][16 * g];
            acc[t] = __builtin_amdgcn_wmma_f32_16x16x32_f16(
                false, afrag, false, bfrag, (short)0, acc[t], false, false);
        }
        __syncthreads();   // all waves done reading buf before it is re-filled
    }

    // epilogue: row = bm+wm+8*(lane>=16)+r, col = bn+wn+16t+(lane&15)
    #pragma unroll
    for (int t = 0; t < 4; t++) {
        int col = bn + wn + t * 16 + lnn;
        if (col >= N) continue;
        #pragma unroll
        for (int r = 0; r < 8; r++) {
            int row = bm + wm + 8 * g + r;
            size_t off = (size_t)row * N + col;
            float v = acc[t][r];
            if (MODE == 0) {
                ((f16*)Cout)[off] = (f16)v;
            } else if (MODE == 1) {
                ((float*)Cout)[off] = v;
            } else {  // softplus(v + bias)
                v += bias[col];
                ((float*)Cout)[off] = (v > 20.f) ? v : logf(1.f + __expf(v));
            }
        }
    }
}

// ---------------------------------------------------------------------------
// depthwise causal conv1d (d_conv=4) + bias + SiLU on xs = xz[:, :1024]
// ---------------------------------------------------------------------------
__global__ __launch_bounds__(256) void dwconv_kernel(
    const f16* __restrict__ xz, const float* __restrict__ w,
    const float* __restrict__ bias, f16* __restrict__ xc) {
    size_t idx = (size_t)blockIdx.x * blockDim.x + threadIdx.x;  // bl*1024+d
    int d = idx & (DINNER - 1);
    size_t bl = idx >> 10;
    int l = (int)(bl & (SEQL - 1));
    size_t row0 = bl - l;  // start-of-sequence row
    float acc = bias[d];
    #pragma unroll
    for (int k = 0; k < 4; k++) {
        int ll = l - 3 + k;
        if (ll >= 0)
            acc += w[d * 4 + k] * (float)xz[(row0 + ll) * (2 * DINNER) + d];
    }
    acc = acc / (1.f + __expf(-acc));  // SiLU
    xc[idx] = (f16)acc;
}

// ---------------------------------------------------------------------------
// slice dt (first 32 cols of xdbl) to f16 for dt_proj GEMM
// ---------------------------------------------------------------------------
__global__ void dt_slice_kernel(const float* __restrict__ xdbl,
                                f16* __restrict__ dt16) {
    size_t idx = (size_t)blockIdx.x * blockDim.x + threadIdx.x;
    if (idx >= (size_t)MROWS * DTRANK) return;
    size_t m = idx >> 5;
    int j = idx & 31;
    dt16[idx] = (f16)xdbl[m * 64 + j];
}

// ---------------------------------------------------------------------------
// selective scan: one thread per (b, d) channel; h[16] in registers.
// B/C per (b,l) broadcast via LDS. Fused: y = (scan + u*D) * silu(z) -> f16
// grid = B*4 blocks of 256 threads
// ---------------------------------------------------------------------------
__global__ __launch_bounds__(256) void scan_kernel(
    const f16* __restrict__ xc, const float* __restrict__ delta,
    const float* __restrict__ xdbl, const f16* __restrict__ xz,
    const float* __restrict__ A_log, const float* __restrict__ Dp,
    f16* __restrict__ y16) {
    int b = blockIdx.x >> 2;
    int d = (blockIdx.x & 3) * 256 + threadIdx.x;
    float h[DSTATE];
    float Aa[DSTATE];
    #pragma unroll
    for (int n = 0; n < DSTATE; n++) {
        h[n]  = 0.f;
        Aa[n] = -__expf(A_log[d * DSTATE + n]);
    }
    float Dd = Dp[d];
    __shared__ float sBC[2 * DSTATE];
    for (int l = 0; l < SEQL; l++) {
        size_t row = (size_t)b * SEQL + l;
        if (threadIdx.x < 2 * DSTATE)
            sBC[threadIdx.x] = xdbl[row * 64 + DTRANK + threadIdx.x];
        __syncthreads();
        float u   = (float)xc[row * DINNER + d];
        float dlt = delta[row * DINNER + d];
        float du  = dlt * u;
        float y   = 0.f;
        #pragma unroll
        for (int n = 0; n < DSTATE; n++) {
            float dA = __expf(dlt * Aa[n]);
            h[n] = dA * h[n] + du * sBC[n];
            y += h[n] * sBC[DSTATE + n];
        }
        float zv = (float)xz[row * (2 * DINNER) + DINNER + d];
        float gate = zv / (1.f + __expf(-zv));
        y16[row * DINNER + d] = (f16)((y + u * Dd) * gate);
        __syncthreads();
    }
}

// ---------------------------------------------------------------------------
// head: last 96 steps, out[b,t,c] = (o @ head_w^T) * std + mean
// ---------------------------------------------------------------------------
__global__ __launch_bounds__(256) void head_kernel(
    const f16* __restrict__ o16, const float* __restrict__ head_w,
    const float* __restrict__ mean, const float* __restrict__ stdv,
    float* __restrict__ out) {
    int row = blockIdx.x;  // 0..767
    int b = row / PREDL, t = row % PREDL;
    size_t m = (size_t)b * SEQL + (SEQL - PREDL) + t;
    __shared__ float sh[DMODEL];
    for (int i = threadIdx.x; i < DMODEL; i += 256)
        sh[i] = (float)o16[m * DMODEL + i];
    __syncthreads();
    int c = threadIdx.x;
    if (c < CIN) {
        float acc = 0.f;
        for (int k = 0; k < DMODEL; k++) acc += sh[k] * head_w[c * DMODEL + k];
        out[(size_t)row * CIN + c] = acc * stdv[b * CIN + c] + mean[b * CIN + c];
    }
}

// ---------------------------------------------------------------------------
extern "C" void kernel_launch(void* const* d_in, const int* in_sizes, int n_in,
                              void* d_out, int out_size, void* d_ws,
                              size_t ws_size, hipStream_t stream) {
    const float* x_enc      = (const float*)d_in[0];
    const float* x_mark_enc = (const float*)d_in[1];
    // d_in[2], d_in[3] (x_dec, x_mark_dec) unused by forecast path
    const float* conv_w    = (const float*)d_in[4];
    const float* temp_w    = (const float*)d_in[5];
    const float* in_proj_w = (const float*)d_in[6];
    const float* conv1d_w  = (const float*)d_in[7];
    const float* conv1d_b  = (const float*)d_in[8];
    const float* x_proj_w  = (const float*)d_in[9];
    const float* dt_proj_w = (const float*)d_in[10];
    const float* dt_proj_b = (const float*)d_in[11];
    const float* A_log     = (const float*)d_in[12];
    const float* Dp        = (const float*)d_in[13];
    const float* out_proj_w = (const float*)d_in[14];
    const float* head_w     = (const float*)d_in[15];
    float* out = (float*)d_out;

    // workspace carve-out
    char* ws = (char*)d_ws;
    size_t off = 0;
    auto alloc = [&](size_t bytes) -> char* {
        char* p = ws + off;
        off = (off + bytes + 255) & ~(size_t)255;
        return p;
    };
    float* mean = (float*)alloc(BATCH * CIN * 4);
    float* istd = (float*)alloc(BATCH * CIN * 4);
    float* stdv = (float*)alloc(BATCH * CIN * 4);
    f16* w16_in  = (f16*)alloc((size_t)2 * DINNER * DMODEL * 2);   // 2048x512
    f16* w16_x   = (f16*)alloc((size_t)64 * DINNER * 2);           // 64x1024
    f16* w16_dt  = (f16*)alloc((size_t)DINNER * DTRANK * 2);       // 1024x32
    f16* w16_out = (f16*)alloc((size_t)DMODEL * DINNER * 2);       // 512x1024
    f16* emb16 = (f16*)alloc((size_t)MROWS * DMODEL * 2);          // 16 MB
    f16* xz16  = (f16*)alloc((size_t)MROWS * 2 * DINNER * 2);      // 64 MB
    f16* xc16  = (f16*)alloc((size_t)MROWS * DINNER * 2);          // 32 MB
    float* xdbl = (float*)alloc((size_t)MROWS * 64 * 4);           // 4 MB
    f16* dt16  = (f16*)alloc((size_t)MROWS * DTRANK * 2);          // 1 MB
    float* delta = (float*)alloc((size_t)MROWS * DINNER * 4);      // 64 MB
    f16* y16   = (f16*)alloc((size_t)MROWS * DINNER * 2);          // 32 MB
    f16* o16   = (f16*)alloc((size_t)MROWS * DMODEL * 2);          // 16 MB
    (void)ws_size;

    // 1) weight conversion f32 -> f16
    {
        size_t n1 = (size_t)2 * DINNER * DMODEL;
        size_t n2 = (size_t)64 * DINNER;
        size_t n3 = (size_t)DINNER * DTRANK;
        size_t n4 = (size_t)DMODEL * DINNER;
        cvt_f32_f16_kernel<<<(n1 + 255) / 256, 256, 0, stream>>>(in_proj_w, w16_in, n1);
        cvt_f32_f16_kernel<<<(n2 + 255) / 256, 256, 0, stream>>>(x_proj_w, w16_x, n2);
        cvt_f32_f16_kernel<<<(n3 + 255) / 256, 256, 0, stream>>>(dt_proj_w, w16_dt, n3);
        cvt_f32_f16_kernel<<<(n4 + 255) / 256, 256, 0, stream>>>(out_proj_w, w16_out, n4);
    }
    // 2) RevIN stats
    stats_kernel<<<BATCH * CIN, 256, 0, stream>>>(x_enc, mean, istd, stdv);
    // 3) embedding
    embed_kernel<<<MROWS, 128, 0, stream>>>(x_enc, x_mark_enc, conv_w, temp_w,
                                            mean, istd, emb16);
    // 4) in_proj GEMM: [16384,512] x [2048,512]^T -> xz16 [16384,2048] f16
    {
        dim3 grid((2 * DINNER + GBN - 1) / GBN, MROWS / GBM);
        gemm_wmma_kernel<0><<<grid, 256, 0, stream>>>(
            emb16, w16_in, nullptr, (void*)xz16, MROWS, 2 * DINNER, DMODEL);
    }
    // 5) depthwise causal conv + SiLU -> xc16
    dwconv_kernel<<<((size_t)MROWS * DINNER) / 256, 256, 0, stream>>>(
        xz16, conv1d_w, conv1d_b, xc16);
    // 6) x_proj GEMM: [16384,1024] x [64,1024]^T -> xdbl [16384,64] f32
    {
        dim3 grid((64 + GBN - 1) / GBN, MROWS / GBM);
        gemm_wmma_kernel<1><<<grid, 256, 0, stream>>>(
            xc16, w16_x, nullptr, (void*)xdbl, MROWS, 64, DINNER);
    }
    // 7) dt slice to f16
    {
        size_t n = (size_t)MROWS * DTRANK;
        dt_slice_kernel<<<(n + 255) / 256, 256, 0, stream>>>(xdbl, dt16);
    }
    // 8) dt_proj GEMM + bias + softplus: [16384,32] x [1024,32]^T -> delta f32
    {
        dim3 grid((DINNER + GBN - 1) / GBN, MROWS / GBM);
        gemm_wmma_kernel<2><<<grid, 256, 0, stream>>>(
            dt16, w16_dt, dt_proj_b, (void*)delta, MROWS, DINNER, DTRANK);
    }
    // 9) selective scan (fused skip + SiLU gate) -> y16
    scan_kernel<<<BATCH * 4, 256, 0, stream>>>(xc16, delta, xdbl, xz16, A_log,
                                               Dp, y16);
    // 10) out_proj GEMM: [16384,1024] x [512,1024]^T -> o16 [16384,512] f16
    {
        dim3 grid((DMODEL + GBN - 1) / GBN, MROWS / GBM);
        gemm_wmma_kernel<0><<<grid, 256, 0, stream>>>(
            y16, w16_out, nullptr, (void*)o16, MROWS, DMODEL, DINNER);
    }
    // 11) head + de-normalize, last 96 steps
    head_kernel<<<BATCH * PREDL, 256, 0, stream>>>(o16, head_w, mean, stdv, out);
}
